// CrfRnn_58609123721967
// MI455X (gfx1250) — matrix operand order repaired and checked by reading, not compiled, not run on hardware
//
#include <hip/hip_runtime.h>

// CRF-as-RNN mean-field for MI455X (gfx1250, wave32, WMMA f16 16x16x32).
// K_sp/K_bi materialized once as fp16 (256 MiB in d_ws); 5 mean-field
// iterations run as fp16 WMMA matmuls streaming K from HBM exactly once per
// iteration. Each wave handles ONE kernel matrix (sp or bi); a 2-deep
// register fragment pipeline is enforced with sched_barrier so the scheduler
// cannot sink next-step loads to their use site (keeps s_wait_loadcnt slack).
// Deterministic K-split (16 chunks) into partial buffers; no float atomics.

typedef __attribute__((ext_vector_type(16))) _Float16 v16h;
typedef __attribute__((ext_vector_type(8)))  _Float16 v8h;
typedef __attribute__((ext_vector_type(8)))  float    v8f;

#define NPIX 8192      // D*H*W = 8*32*32
#define NLAB 21
#define LPAD 32        // pad labels to 2 WMMA M-tiles
#define KCHUNKS 16     // K-dim split (8192/16 = 512 per chunk = 16 WMMA steps)
#define KSTEPS  16     // K-steps of 32 per chunk

// ---------------------------------------------------------------------------
// 1) Build K_sp and K_bi (fp16, row-major NxN). Symmetric, so later we can
//    read B-fragments with contiguous K. Pure VALU + v_exp_f32.
// ---------------------------------------------------------------------------
__global__ void crf_build_kernels(const float* __restrict__ img,
                                  _Float16* __restrict__ Ksp,
                                  _Float16* __restrict__ Kbi) {
  const int N = NPIX;
  __shared__ float fj[6][256];
  const int jb = (blockIdx.x & 31) << 8;   // 32 j-blocks of 256
  const int ib = (blockIdx.x >> 5) << 5;   // 256 i-blocks of 32
  const int t  = threadIdx.x;
  const int j  = jb + t;

  fj[0][t] = (float)(j >> 10);
  fj[1][t] = (float)((j >> 5) & 31);
  fj[2][t] = (float)(j & 31);
  fj[3][t] = img[j];
  fj[4][t] = img[N + j];
  fj[5][t] = img[2 * N + j];
  __syncthreads();

  const float invG2 = 1.0f / (3.0f * 3.0f);       // gamma^2
  const float invA2 = 1.0f / (160.0f * 160.0f);   // alpha^2
  const float invB2 = 1.0f / (3.0f * 3.0f);       // beta^2

  for (int ii = 0; ii < 32; ++ii) {
    const int i = ib + ii;
    const float zi = (float)(i >> 10);
    const float yi = (float)((i >> 5) & 31);
    const float xi = (float)(i & 31);
    const float r = img[i], g = img[N + i], b = img[2 * N + i];  // broadcast reads
    const float dz = zi - fj[0][t], dy = yi - fj[1][t], dx = xi - fj[2][t];
    const float dr = r - fj[3][t],  dg = g - fj[4][t], db = b - fj[5][t];
    const float d2p = dz * dz + dy * dy + dx * dx;
    const float d2c = dr * dr + dg * dg + db * db;
    Ksp[(size_t)i * N + j] = (_Float16)__expf(-0.5f * d2p * invG2);
    Kbi[(size_t)i * N + j] = (_Float16)__expf(-0.5f * (d2p * invA2 + d2c * invB2));
  }
}

// ---------------------------------------------------------------------------
// 2) Row sums (== column sums by symmetry): normalizers n_sp, n_bi.
// ---------------------------------------------------------------------------
__global__ void crf_row_sums(const _Float16* __restrict__ Ksp,
                             const _Float16* __restrict__ Kbi,
                             float* __restrict__ nsp, float* __restrict__ nbi) {
  __shared__ float s0[256], s1[256];
  const int row = blockIdx.x;
  const int t = threadIdx.x;
  const _Float16* a = Ksp + (size_t)row * NPIX;
  const _Float16* b = Kbi + (size_t)row * NPIX;
  float sa = 0.f, sb = 0.f;
  for (int c = t * 8; c < NPIX; c += 256 * 8) {
    v8h va = *(const v8h*)(a + c);
    v8h vb = *(const v8h*)(b + c);
#pragma unroll
    for (int e = 0; e < 8; ++e) { sa += (float)va[e]; sb += (float)vb[e]; }
  }
  s0[t] = sa; s1[t] = sb;
  __syncthreads();
  for (int off = 128; off > 0; off >>= 1) {
    if (t < off) { s0[t] += s0[t + off]; s1[t] += s1[t + off]; }
    __syncthreads();
  }
  if (t == 0) { nsp[row] = s0[0]; nbi[row] = s1[0]; }
}

// ---------------------------------------------------------------------------
// 3) cur = logits (f32 copy), and precombined label matrices C@Wsp, C@Wbi.
// ---------------------------------------------------------------------------
__global__ void crf_init_cur(const float* __restrict__ logits, float* __restrict__ cur) {
  const int i = blockIdx.x * 256 + threadIdx.x;
  if (i < NLAB * NPIX) cur[i] = logits[i];
}

__global__ void crf_combine(const float* __restrict__ Wsp, const float* __restrict__ Wbi,
                            const float* __restrict__ Cm,
                            float* __restrict__ WcSp, float* __restrict__ WcBi) {
  const int idx = blockIdx.x * 64 + threadIdx.x;
  if (idx >= NLAB * NLAB) return;
  const int l = idx / NLAB, p = idx % NLAB;
  float a = 0.f, b = 0.f;
  for (int m = 0; m < NLAB; ++m) {
    a += Cm[l * NLAB + m] * Wsp[m * NLAB + p];
    b += Cm[l * NLAB + m] * Wbi[m * NLAB + p];
  }
  WcSp[idx] = a; WcBi[idx] = b;
}

// ---------------------------------------------------------------------------
// 4) q = softmax(cur) over labels, stored fp16 padded to 32 rows (21..31 = 0).
// ---------------------------------------------------------------------------
__global__ void crf_softmax_q(const float* __restrict__ cur, _Float16* __restrict__ q) {
  const int j = blockIdx.x * 256 + threadIdx.x;
  float v[NLAB];
  float m = -1e30f;
#pragma unroll
  for (int l = 0; l < NLAB; ++l) { v[l] = cur[l * NPIX + j]; m = fmaxf(m, v[l]); }
  float s = 0.f;
#pragma unroll
  for (int l = 0; l < NLAB; ++l) { v[l] = __expf(v[l] - m); s += v[l]; }
  const float inv = 1.0f / s;
#pragma unroll
  for (int l = 0; l < NLAB; ++l) q[l * NPIX + j] = (_Float16)(v[l] * inv);
#pragma unroll
  for (int l = NLAB; l < LPAD; ++l) q[l * NPIX + j] = (_Float16)0.0f;
}

// ---------------------------------------------------------------------------
// 5) WMMA matmul: partial[kc] = q(32xN) @ Km(NxN) for ONE kernel matrix per
//    wave (wave's top id bit selects Ksp or Kbi; K still streamed once total).
//    Per wave: 32x32 output tile, K-chunk of 512, 4 v_wmma per K-step of 32,
//    fully-unrolled 2-deep register pipeline enforced with sched_barrier.
//    Fragment packing per ISA 7.12.2 (wave32):
//      A 16x32 f16 : row M = lane%16; halves 0..7 -> K = 8*(lane/16)+h,
//                    halves 8..15 -> K = 16 + 8*(lane/16) + (h-8)
//      B 32x16 f16 : col N = lane%16; same K mapping as A
//                    (K is symmetric, so B[k][n] = Kmat[n_row][k] contiguous)
//      C/D f32     : VGPR r -> M = r + 8*(lane/16), N = lane%16
// ---------------------------------------------------------------------------
__device__ __forceinline__ v16h ld_a_frag(const _Float16* p) {
  v8h a0 = *(const v8h*)p;          // K = lo .. lo+7
  v8h a1 = *(const v8h*)(p + 16);   // K = lo+16 .. lo+23
  v16h a;
#pragma unroll
  for (int e = 0; e < 8; ++e) { a[e] = a0[e]; a[e + 8] = a1[e]; }
  return a;
}
__device__ __forceinline__ v16h ld_b_frag(const _Float16* p) {
  v8h b0 = *(const v8h*)p;          // K = base .. base+7
  v8h b1 = *(const v8h*)(p + 16);   // K = base+16 .. base+23
  v16h b;
#pragma unroll
  for (int e = 0; e < 8; ++e) { b[e] = b0[e]; b[e + 8] = b1[e]; }
  return b;
}

struct Frags {
  v16h a0, a1;   // A: M-tile 0 (rows 0..15), M-tile 1 (rows 16..31)
  v16h b0, b1;   // B: n-tile 0, n-tile 1
};

__global__ void __launch_bounds__(128)
crf_matmul(const _Float16* __restrict__ q,
           const _Float16* __restrict__ Ksp,
           const _Float16* __restrict__ Kbi,
           float* __restrict__ psp, float* __restrict__ pbi) {
  const int N = NPIX;
  const int wid  = (blockIdx.x * 128 + (int)threadIdx.x) >> 5;
  const int lane = threadIdx.x & 31;
  const int nt  = wid & 255;           // 256 column tiles of 32
  const int kc  = (wid >> 8) & 15;     // 16 K-chunks of 512
  const int mat = wid >> 12;           // 0 = Ksp/psp, 1 = Kbi/pbi
  const int jbase = nt << 5;
  const int lrow = lane & 15;
  const int lhi  = lane >> 4;

  const _Float16* Km  = mat ? Kbi : Ksp;
  float*          pot = mat ? pbi : psp;

  // acc[mt][ntile]
  v8f acc[2][2] = {};

  const _Float16* aP0 = q  + (size_t)lrow * N               + (lhi << 3);
  const _Float16* aP1 = q  + (size_t)(16 + lrow) * N        + (lhi << 3);
  const _Float16* bP0 = Km + (size_t)(jbase + lrow) * N      + (lhi << 3);
  const _Float16* bP1 = Km + (size_t)(jbase + 16 + lrow) * N + (lhi << 3);

  auto load_frags = [&](Frags& f, int k) {
    f.a0 = ld_a_frag(aP0 + k);
    f.a1 = ld_a_frag(aP1 + k);
    f.b0 = ld_b_frag(bP0 + k);
    f.b1 = ld_b_frag(bP1 + k);
  };
  auto do_wmmas = [&](const Frags& f) {
    acc[0][0] = __builtin_amdgcn_wmma_f32_16x16x32_f16(false, f.a0, false, f.b0, (short)0, acc[0][0], false, false);
    acc[0][1] = __builtin_amdgcn_wmma_f32_16x16x32_f16(false, f.a0, false, f.b1, (short)0, acc[0][1], false, false);
    acc[1][0] = __builtin_amdgcn_wmma_f32_16x16x32_f16(false, f.a1, false, f.b0, (short)0, acc[1][0], false, false);
    acc[1][1] = __builtin_amdgcn_wmma_f32_16x16x32_f16(false, f.a1, false, f.b1, (short)0, acc[1][1], false, false);
  };

  const int kBeg = kc << 9;   // 512 K per chunk
  Frags buf[2];
  load_frags(buf[0], kBeg);
#pragma unroll
  for (int ks = 0; ks < KSTEPS; ++ks) {
    if (ks + 1 < KSTEPS) {
      const int kN = kBeg + ((ks + 1) << 5);
      // Warm L2 ahead of the stream (global_prefetch_b8); all 32 lanes cover
      // the 32 rows of this wave's column tile.
      __builtin_prefetch(bP0 + kN + 256, 0, 1);
      __builtin_prefetch(bP1 + kN + 256, 0, 1);
      load_frags(buf[(ks + 1) & 1], kN);   // compile-time buffer index
    }
    // Pin the software pipeline: next-step loads must ISSUE before this
    // step's WMMAs (scheduler may not sink them to their use site).
    __builtin_amdgcn_sched_barrier(0);
    do_wmmas(buf[ks & 1]);
  }

  // Write per-chunk partials (every [kc][row<32][col] written exactly once
  // per matrix).
  float* o = pot + (size_t)kc * LPAD * N;
#pragma unroll
  for (int mt = 0; mt < 2; ++mt) {
#pragma unroll
    for (int t = 0; t < 2; ++t) {
      const int col = jbase + (t << 4) + lrow;
#pragma unroll
      for (int r = 0; r < 8; ++r) {
        const int row = (mt << 4) + r + (lhi << 3);
        o[(size_t)row * N + col] = acc[mt][t][r];
      }
    }
  }
}

// ---------------------------------------------------------------------------
// 6) Reduce K-chunks, normalize, apply label mixing, cur = un + msg.
// ---------------------------------------------------------------------------
__global__ void crf_update(const float* __restrict__ psp, const float* __restrict__ pbi,
                           const float* __restrict__ nsp, const float* __restrict__ nbi,
                           const float* __restrict__ un,
                           const float* __restrict__ WcSp, const float* __restrict__ WcBi,
                           float* __restrict__ cur) {
  __shared__ float sA[NLAB * NLAB], sB[NLAB * NLAB];
  for (int i = threadIdx.x; i < NLAB * NLAB; i += 256) {
    sA[i] = WcSp[i]; sB[i] = WcBi[i];
  }
  __syncthreads();

  const int j = blockIdx.x * 256 + threadIdx.x;
  const float isp = 1.0f / nsp[j];
  const float ibi = 1.0f / nbi[j];

  float sp[NLAB], bi[NLAB];
#pragma unroll
  for (int l = 0; l < NLAB; ++l) {
    float a = 0.f, b = 0.f;
#pragma unroll
    for (int kc = 0; kc < KCHUNKS; ++kc) {
      const size_t o = ((size_t)kc * LPAD + l) * NPIX + j;
      a += psp[o]; b += pbi[o];
    }
    sp[l] = a * isp;
    bi[l] = b * ibi;
  }
#pragma unroll
  for (int l = 0; l < NLAB; ++l) {
    float s = un[l * NPIX + j];
#pragma unroll
    for (int p = 0; p < NLAB; ++p)
      s += sA[l * NLAB + p] * sp[p] + sB[l * NLAB + p] * bi[p];
    cur[l * NPIX + j] = s;
  }
}

// ---------------------------------------------------------------------------
// 7) Final softmax -> f32 output (1, L, D, H, W) row-major.
// ---------------------------------------------------------------------------
__global__ void crf_out(const float* __restrict__ cur, float* __restrict__ out) {
  const int j = blockIdx.x * 256 + threadIdx.x;
  float v[NLAB];
  float m = -1e30f;
#pragma unroll
  for (int l = 0; l < NLAB; ++l) { v[l] = cur[l * NPIX + j]; m = fmaxf(m, v[l]); }
  float s = 0.f;
#pragma unroll
  for (int l = 0; l < NLAB; ++l) { v[l] = __expf(v[l] - m); s += v[l]; }
  const float inv = 1.0f / s;
#pragma unroll
  for (int l = 0; l < NLAB; ++l) out[l * NPIX + j] = v[l] * inv;
}

// ---------------------------------------------------------------------------
// Workspace layout (bytes, all offsets 256-aligned). Total ~289.2 MiB.
// ---------------------------------------------------------------------------
#define WS_KSP   0ull
#define WS_KBI   134217728ull                        // 8192*8192*2
#define WS_NSP   268435456ull                        // 8192*4
#define WS_NBI   268468224ull
#define WS_CUR   268500992ull                        // 21*8192*4
#define WS_Q     269189120ull                        // 32*8192*2
#define WS_PSP   269713408ull                        // 16*32*8192*4 = 16 MiB
#define WS_PBI   286490624ull                        // 16*32*8192*4 = 16 MiB
#define WS_WCSP  303267840ull                        // 441*4
#define WS_WCBI  303269888ull

extern "C" void kernel_launch(void* const* d_in, const int* in_sizes, int n_in,
                              void* d_out, int out_size, void* d_ws, size_t ws_size,
                              hipStream_t stream) {
  const float* img    = (const float*)d_in[0];
  const float* logits = (const float*)d_in[1];
  const float* unary  = (const float*)d_in[2];
  const float* Wsp    = (const float*)d_in[3];
  const float* Wbi    = (const float*)d_in[4];
  const float* Cm     = (const float*)d_in[5];
  float* out = (float*)d_out;

  char* ws = (char*)d_ws;
  _Float16* Ksp = (_Float16*)(ws + WS_KSP);
  _Float16* Kbi = (_Float16*)(ws + WS_KBI);
  float*    nsp = (float*)(ws + WS_NSP);
  float*    nbi = (float*)(ws + WS_NBI);
  float*    cur = (float*)(ws + WS_CUR);
  _Float16* qm  = (_Float16*)(ws + WS_Q);
  float*    psp = (float*)(ws + WS_PSP);
  float*    pbi = (float*)(ws + WS_PBI);
  float*    WcSp = (float*)(ws + WS_WCSP);
  float*    WcBi = (float*)(ws + WS_WCBI);

  crf_build_kernels<<<8192, 256, 0, stream>>>(img, Ksp, Kbi);
  crf_row_sums<<<NPIX, 256, 0, stream>>>(Ksp, Kbi, nsp, nbi);
  crf_init_cur<<<(NLAB * NPIX + 255) / 256, 256, 0, stream>>>(logits, cur);
  crf_combine<<<7, 64, 0, stream>>>(Wsp, Wbi, Cm, WcSp, WcBi);

  for (int it = 0; it < 5; ++it) {
    crf_softmax_q<<<NPIX / 256, 256, 0, stream>>>(cur, qm);
    // 8192 waves: 256 nt x 16 kc x 2 matrices, 4 waves per block.
    crf_matmul<<<2048, 128, 0, stream>>>(qm, Ksp, Kbi, psp, pbi);
    crf_update<<<NPIX / 256, 256, 0, stream>>>(psp, pbi, nsp, nbi, unary,
                                               WcSp, WcBi, cur);
  }
  crf_out<<<NPIX / 256, 256, 0, stream>>>(cur, out);
}
